// TetConv_80221399154840
// MI455X (gfx1250) — matrix operand order repaired and checked by reading, not compiled, not run on hardware
//
#include <hip/hip_runtime.h>
#include <hip/hip_bf16.h>

// ---------- types for WMMA ----------
typedef __bf16 bf16_t;
typedef __attribute__((ext_vector_type(16))) __bf16 v16bf;
typedef __attribute__((ext_vector_type(8)))  float  v8f;

struct alignas(16) U128 { unsigned int x, y, z, w; };   // POD 16B chunk

#define THREADS 256

// ---------------------------------------------------------------------------
// small elementwise kernels
// ---------------------------------------------------------------------------
__global__ void set_const_kernel(float* __restrict__ p, float v, long n) {
    long i = (long)blockIdx.x * blockDim.x + threadIdx.x;
    if (i < n) p[i] = v;
}

// deg[v] += 3 for each of the 4 vertices of each tet (each vertex is dst of 3
// directed edges per tet); self-loop contributes the initial 1.
__global__ void tet_deg_kernel(const int* __restrict__ tets, float* __restrict__ deg,
                               int n_tets) {
    long t = (long)blockIdx.x * blockDim.x + threadIdx.x;
    if (t >= n_tets) return;
    const int* q = tets + 4 * t;
    __builtin_prefetch(q + 4 * THREADS, 0, 0);   // global_prefetch_b8: next block's rows
    atomicAdd(deg + q[0], 3.0f);
    atomicAdd(deg + q[1], 3.0f);
    atomicAdd(deg + q[2], 3.0f);
    atomicAdd(deg + q[3], 3.0f);
}

__global__ void rsqrt_inplace_kernel(float* __restrict__ p, long n) {
    long i = (long)blockIdx.x * blockDim.x + threadIdx.x;
    if (i < n) p[i] = rsqrtf(p[i]);     // deg >= 1 always
}

// f32 -> bf16 convert, optional leaky-relu(0.01), zero-pad to n_total
__global__ void cvt_bf16_kernel(const float* __restrict__ src, bf16_t* __restrict__ dst,
                                long n_src, long n_total, int apply_nl) {
    long i = (long)blockIdx.x * blockDim.x + threadIdx.x;
    if (i >= n_total) return;
    float v = 0.0f;
    if (i < n_src) {
        v = src[i];
        if (apply_nl && v < 0.0f) v *= 0.01f;
    }
    dst[i] = (bf16_t)v;
}

// x_vert [N,3] @ w_in1^T [3,256] + b_in1  (K=3: VALU is the right tool)
__global__ void lin_in1_kernel(const float* __restrict__ x, const float* __restrict__ w,
                               const float* __restrict__ b, float* __restrict__ out, int n) {
    long tid = (long)blockIdx.x * blockDim.x + threadIdx.x;
    if (tid >= (long)n * 256) return;
    int i = (int)(tid >> 8);
    int j = (int)(tid & 255);
    const float* wr = w + 3 * j;
    const float* xr = x + 3 * i;
    out[tid] = b[j] + xr[0] * wr[0] + xr[1] * wr[1] + xr[2] * wr[2];
}

// Edge scatter: for each tet, 12 directed edges; z[d,f] += dinv[s]*dinv[d]*y[s,f]
__global__ void scatter_kernel(const int* __restrict__ tets, const float* __restrict__ y,
                               const float* __restrict__ dinv, float* __restrict__ z,
                               int n_tets, int dout) {
    long tid = (long)blockIdx.x * blockDim.x + threadIdx.x;
    if (tid >= (long)n_tets * dout) return;
    int t = (int)(tid / dout);
    int f = (int)(tid - (long)t * dout);
    const int* q = tets + 4 * t;
    int v0 = q[0], v1 = q[1], v2 = q[2], v3 = q[3];
    float d0 = dinv[v0], d1 = dinv[v1], d2 = dinv[v2], d3 = dinv[v3];
    float y0 = y[(size_t)v0 * dout + f];
    float y1 = y[(size_t)v1 * dout + f];
    float y2 = y[(size_t)v2 * dout + f];
    float y3 = y[(size_t)v3 * dout + f];
    float* z0 = z + (size_t)v0 * dout + f;
    float* z1 = z + (size_t)v1 * dout + f;
    float* z2 = z + (size_t)v2 * dout + f;
    float* z3 = z + (size_t)v3 * dout + f;
    // unordered pairs {0,1},{0,2},{0,3},{1,2},{1,3},{2,3}, both directions
    atomicAdd(z1, d0 * d1 * y0);  atomicAdd(z0, d0 * d1 * y1);
    atomicAdd(z2, d0 * d2 * y0);  atomicAdd(z0, d0 * d2 * y2);
    atomicAdd(z3, d0 * d3 * y0);  atomicAdd(z0, d0 * d3 * y3);
    atomicAdd(z2, d1 * d2 * y1);  atomicAdd(z1, d1 * d2 * y2);
    atomicAdd(z3, d1 * d3 * y1);  atomicAdd(z1, d1 * d3 * y3);
    atomicAdd(z3, d2 * d3 * y2);  atomicAdd(z2, d2 * d3 * y3);
}

__global__ void residual_kernel(float* __restrict__ x, const float* __restrict__ h, long n) {
    long i = (long)blockIdx.x * blockDim.x + threadIdx.x;
    if (i < n) x[i] = (x[i] + h[i]) * 0.70710678118654752f;
}

// h8 [N,8] -> d_out = concat(h[:, :3] flat, h[:,3], h[:,4:] flat)
__global__ void split_kernel(const float* __restrict__ h8, float* __restrict__ out, int n) {
    long i = (long)blockIdx.x * blockDim.x + threadIdx.x;
    if (i >= n) return;
    const float* r = h8 + 8 * i;
    out[3 * i + 0] = r[0];
    out[3 * i + 1] = r[1];
    out[3 * i + 2] = r[2];
    out[(size_t)3 * n + i] = r[3];
    float* o2 = out + (size_t)4 * n + 4 * i;
    o2[0] = r[4]; o2[1] = r[5]; o2[2] = r[6]; o2[3] = r[7];
}

// ---------------------------------------------------------------------------
// WMMA GEMM: C[M,Nout] = A[M,K](bf16) @ W^T, W[Nout,K](bf16) row-major.
// Each wave computes a 16 x (NT*16) tile: A fragment held in registers and
// reused across NT WMMAs per K-step (NT=4 for the wide layers).
// Optional fused epilogue: Z[i,f] = dinv[i]^2 * C[i,f] + zbias[f]
// (GCN self-loop + conv-bias init, saves a full elementwise pass).
// M multiple of 16 (50000 = 3125*16). Nout=8 handled via padded W + store
// guard; the guard diverges only AFTER all WMMAs (EXEC all-1s at WMMA time).
// ---------------------------------------------------------------------------
template <int K, int NT>
__global__ __launch_bounds__(256) void gemm_bf16_wmma(
        const bf16_t* __restrict__ A, const bf16_t* __restrict__ B,
        float* __restrict__ C, const float* __restrict__ bias,
        float* __restrict__ Z, const float* __restrict__ dinv,
        const float* __restrict__ zbias,
        int Nout, int nGroupN, int nTiles) {
    const int wave = threadIdx.x >> 5;            // 8 waves / block
    const int lane = threadIdx.x & 31;
    const int tile = blockIdx.x * 8 + wave;       // wave-uniform
    if (tile >= nTiles) return;                   // wave-uniform exit

    const int tm = tile / nGroupN;
    const int tg = tile - tm * nGroupN;
    const int half = lane >> 4;                   // 0: lanes 0-15, 1: lanes 16-31
    const int idx  = lane & 15;

    // A lane base: row tm*16+idx, K offset half*8 (ISA 16-bit A 16x32 layout)
    const bf16_t* pa = A + (size_t)(tm * 16 + idx) * K + half * 8;
    // B lane base per N-subtile: column of W^T == row of W, K offset half*8
    const bf16_t* pb[NT];
#pragma unroll
    for (int nt = 0; nt < NT; ++nt)
        pb[nt] = B + (size_t)((tg * NT + nt) * 16 + idx) * K + half * 8;

    v8f acc[NT] = {};
#pragma unroll
    for (int k0 = 0; k0 < K; k0 += 32) {
        v16bf af;
        // A frag: K {k0+8h..+7} in v0-3, {k0+16+8h..+7} in v4-7
        ((U128*)&af)[0] = *(const U128*)(pa + k0);
        ((U128*)&af)[1] = *(const U128*)(pa + k0 + 16);
#pragma unroll
        for (int nt = 0; nt < NT; ++nt) {
            v16bf bfr;                            // K {k0+8h..+7}, {k0+16+8h..+7}
            ((U128*)&bfr)[0] = *(const U128*)(pb[nt] + k0);
            ((U128*)&bfr)[1] = *(const U128*)(pb[nt] + k0 + 16);
            acc[nt] = __builtin_amdgcn_wmma_f32_16x16x32_bf16(
                          false, af, false, bfr, (short)0, acc[nt], false, false);
        }
    }

#pragma unroll
    for (int nt = 0; nt < NT; ++nt) {
        const int col = (tg * NT + nt) * 16 + idx;
        if (col < Nout) {                         // only diverges for Nout=8 tail
            float badd = bias ? bias[col] : 0.0f;
            float zb = zbias ? zbias[col] : 0.0f;
#pragma unroll
            for (int r = 0; r < 8; ++r) {         // C/D: VGPR r -> M = r + 8*half
                int row = tm * 16 + r + half * 8;
                float v = acc[nt][r] + badd;
                C[(size_t)row * Nout + col] = v;
                if (Z) {                          // fused self-loop + conv bias
                    float di = dinv[row];
                    Z[(size_t)row * Nout + col] = di * di * v + zb;
                }
            }
        }
    }
}

// ---------------------------------------------------------------------------
extern "C" void kernel_launch(void* const* d_in, const int* in_sizes, int n_in,
                              void* d_out, int out_size, void* d_ws, size_t ws_size,
                              hipStream_t stream) {
    const float* x_vert = (const float*)d_in[0];
    const int*   tets   = (const int*)  d_in[1];
    const float* w_in1 = (const float*)d_in[2];  const float* b_in1 = (const float*)d_in[3];
    const float* wg1a  = (const float*)d_in[4];  const float* bg1a  = (const float*)d_in[5];
    const float* wg1b  = (const float*)d_in[6];  const float* bg1b  = (const float*)d_in[7];
    const float* w_in2 = (const float*)d_in[8];  const float* b_in2 = (const float*)d_in[9];
    const float* wg2a  = (const float*)d_in[10]; const float* bg2a  = (const float*)d_in[11];
    const float* wg2b  = (const float*)d_in[12]; const float* bg2b  = (const float*)d_in[13];
    const float* w_out = (const float*)d_in[14]; const float* b_out = (const float*)d_in[15];
    const float* w_l1  = (const float*)d_in[16]; const float* b_l1  = (const float*)d_in[17];
    const float* w_l2  = (const float*)d_in[18]; const float* b_l2  = (const float*)d_in[19];

    const int N      = in_sizes[0] / 3;     // 50000 (multiple of 16)
    const int n_tets = in_sizes[1] / 4;     // 100000

    // ---- carve workspace (256B aligned) ----
    char* ws = (char*)d_ws;
    size_t off = 0;
    auto carve = [&](size_t bytes) -> void* {
        off = (off + 255) & ~(size_t)255;
        void* p = ws + off;
        off += bytes;
        return p;
    };
    float*  f0     = (float*)carve((size_t)N * 256 * sizeof(float));
    float*  f1     = (float*)carve((size_t)N * 256 * sizeof(float));
    float*  f2     = (float*)carve((size_t)N * 256 * sizeof(float));
    bf16_t* a16    = (bf16_t*)carve((size_t)N * 256 * sizeof(bf16_t));
    float*  dinv   = (float*)carve((size_t)N * sizeof(float));
    bf16_t* bw_g1a = (bf16_t*)carve((size_t)256 * 256 * 2);
    bf16_t* bw_g1b = (bf16_t*)carve((size_t)256 * 256 * 2);
    bf16_t* bw_in2 = (bf16_t*)carve((size_t)128 * 256 * 2);
    bf16_t* bw_g2a = (bf16_t*)carve((size_t)128 * 128 * 2);
    bf16_t* bw_g2b = (bf16_t*)carve((size_t)128 * 128 * 2);
    bf16_t* bw_out = (bf16_t*)carve((size_t)128 * 128 * 2);
    bf16_t* bw_l1  = (bf16_t*)carve((size_t)64 * 128 * 2);
    bf16_t* bw_l2  = (bf16_t*)carve((size_t)16 * 64 * 2);   // padded 8->16 rows

    auto nb = [](long n) { return (unsigned)((n + THREADS - 1) / THREADS); };

    auto cvt = [&](const float* s, bf16_t* d, long n_src, long n_tot, int nl) {
        cvt_bf16_kernel<<<nb(n_tot), THREADS, 0, stream>>>(s, d, n_src, n_tot, nl);
    };

    // K=256, 16x64 tile/wave
    auto gemm256 = [&](const bf16_t* A, const bf16_t* B, float* C, const float* bias,
                       float* Z, const float* zb, int Nout) {
        int nGroupN = (Nout + 63) / 64;
        int nTiles = (N / 16) * nGroupN;
        gemm_bf16_wmma<256, 4><<<(unsigned)((nTiles + 7) / 8), 256, 0, stream>>>(
            A, B, C, bias, Z, dinv, zb, Nout, nGroupN, nTiles);
    };
    // K=128, 16x64 tile/wave
    auto gemm128 = [&](const bf16_t* A, const bf16_t* B, float* C, const float* bias,
                       float* Z, const float* zb, int Nout) {
        int nGroupN = (Nout + 63) / 64;
        int nTiles = (N / 16) * nGroupN;
        gemm_bf16_wmma<128, 4><<<(unsigned)((nTiles + 7) / 8), 256, 0, stream>>>(
            A, B, C, bias, Z, dinv, zb, Nout, nGroupN, nTiles);
    };
    // K=64, 16x16 tile/wave (final Nout=8 layer, padded weights + store guard)
    auto gemm64 = [&](const bf16_t* A, const bf16_t* B, float* C, const float* bias,
                      int Nout) {
        int nGroupN = (Nout + 15) / 16;
        int nTiles = (N / 16) * nGroupN;
        gemm_bf16_wmma<64, 1><<<(unsigned)((nTiles + 7) / 8), 256, 0, stream>>>(
            A, B, C, bias, nullptr, nullptr, nullptr, Nout, nGroupN, nTiles);
    };

    // ---- degree / normalization: deg = 1 (self loop) + 3 per tet incidence ----
    set_const_kernel<<<nb(N), THREADS, 0, stream>>>(dinv, 1.0f, N);
    tet_deg_kernel<<<nb(n_tets), THREADS, 0, stream>>>(tets, dinv, n_tets);
    rsqrt_inplace_kernel<<<nb(N), THREADS, 0, stream>>>(dinv, N);

    // ---- weights -> bf16 (once per call; deterministic) ----
    cvt(wg1a, bw_g1a, 256 * 256, 256 * 256, 0);
    cvt(wg1b, bw_g1b, 256 * 256, 256 * 256, 0);
    cvt(w_in2, bw_in2, 128 * 256, 128 * 256, 0);
    cvt(wg2a, bw_g2a, 128 * 128, 128 * 128, 0);
    cvt(wg2b, bw_g2b, 128 * 128, 128 * 128, 0);
    cvt(w_out, bw_out, 128 * 128, 128 * 128, 0);
    cvt(w_l1, bw_l1, 64 * 128, 64 * 128, 0);
    cvt(w_l2, bw_l2, 8 * 64, 16 * 64, 0);        // zero-pad rows 8..15

    // ---- GCN block 1 (d=256) ----
    lin_in1_kernel<<<nb((long)N * 256), THREADS, 0, stream>>>(x_vert, w_in1, b_in1, f0, N);  // x0

    cvt(f0, a16, (long)N * 256, (long)N * 256, 1);                 // nl(x0)
    gemm256(a16, bw_g1a, f1, nullptr, f2, bg1a, 256);              // y->f1, z-init->f2
    scatter_kernel<<<nb((long)n_tets * 256), THREADS, 0, stream>>>(tets, f1, dinv, f2, n_tets, 256);

    cvt(f2, a16, (long)N * 256, (long)N * 256, 1);                 // nl(h1)
    gemm256(a16, bw_g1b, f1, nullptr, f2, bg1b, 256);
    scatter_kernel<<<nb((long)n_tets * 256), THREADS, 0, stream>>>(tets, f1, dinv, f2, n_tets, 256);

    residual_kernel<<<nb((long)N * 256), THREADS, 0, stream>>>(f0, f2, (long)N * 256);  // x1 in f0

    // ---- GCN block 2 (d=128) ----
    cvt(f0, a16, (long)N * 256, (long)N * 256, 1);                 // nl(x1)
    gemm256(a16, bw_in2, f1, b_in2, nullptr, nullptr, 128);        // x2 in f1

    cvt(f1, a16, (long)N * 128, (long)N * 128, 1);
    gemm128(a16, bw_g2a, f2, nullptr, f0, bg2a, 128);              // y->f2, z-init->f0
    scatter_kernel<<<nb((long)n_tets * 128), THREADS, 0, stream>>>(tets, f2, dinv, f0, n_tets, 128);

    cvt(f0, a16, (long)N * 128, (long)N * 128, 1);
    gemm128(a16, bw_g2b, f2, nullptr, f0, bg2b, 128);
    scatter_kernel<<<nb((long)n_tets * 128), THREADS, 0, stream>>>(tets, f2, dinv, f0, n_tets, 128);

    residual_kernel<<<nb((long)N * 128), THREADS, 0, stream>>>(f1, f0, (long)N * 128);  // x3 in f1

    // ---- MLP head ----
    cvt(f1, a16, (long)N * 128, (long)N * 128, 0);                 // NO nl before w_out
    gemm128(a16, bw_out, f2, b_out, nullptr, nullptr, 128);

    cvt(f2, a16, (long)N * 128, (long)N * 128, 1);
    gemm128(a16, bw_l1, f0, b_l1, nullptr, nullptr, 64);

    cvt(f0, a16, (long)N * 64, (long)N * 64, 1);
    gemm64(a16, bw_l2, f2, b_l2, 8);                               // [N,8] in f2

    split_kernel<<<nb(N), THREADS, 0, stream>>>(f2, (float*)d_out, N);
}
